// ThalamicQuantumEngine_47098611368142
// MI455X (gfx1250) — matrix operand order repaired and checked by reading, not compiled, not run on hardware
//
#include <hip/hip_runtime.h>

// ---------------- types ----------------
typedef __attribute__((ext_vector_type(16))) __bf16 v16bf;
typedef __attribute__((ext_vector_type(8)))  __bf16 v8bf;
typedef __attribute__((ext_vector_type(8)))  float  v8f;

#define NH   64
#define NC   1984
#define INN  1024
#define HH   2048
#define OUTN 1024
#define BM   128

// ---------------- helpers ----------------
__device__ __forceinline__ float wave_sum(float v) {
  #pragma unroll
  for (int off = 16; off > 0; off >>= 1) v += __shfl_xor(v, off, 32);
  return v;
}
__device__ __forceinline__ float wave_max(float v) {
  #pragma unroll
  for (int off = 16; off > 0; off >>= 1) v = fmaxf(v, __shfl_xor(v, off, 32));
  return v;
}
// fast sigmoid / tanh: v_exp_f32 + v_rcp_f32, no IEEE div chains
__device__ __forceinline__ float sigm(float v) {
  return __builtin_amdgcn_rcpf(1.0f + __expf(-v));
}
__device__ __forceinline__ float tanh_fast(float v) {
  return 1.0f - 2.0f * __builtin_amdgcn_rcpf(__expf(2.0f * v) + 1.0f);
}

// A fragment: bf16 row-major [M,K]; 16x32 tile at (tm,k0).
// lane<16: M=lane, K = k0+{0..7, 16..23}; lane>=16: M=lane-16, K = k0+{8..15, 24..31}
__device__ __forceinline__ v16bf load_a_frag(const __bf16* __restrict__ A, int K,
                                             int tm, int k0, int lane) {
  int m  = tm + (lane & 15);
  int hi = (lane >> 4) << 3;                // 0 or 8
  const __bf16* p = A + (size_t)m * K + k0 + hi;
  v8bf c0 = *(const v8bf*)(p);
  v8bf c1 = *(const v8bf*)(p + 16);
  return __builtin_shufflevector(c0, c1, 0,1,2,3,4,5,6,7,8,9,10,11,12,13,14,15);
}

// B fragment: fp32 row-major W[N,K]; B[k][n] = W[n][k]. 32x16 tile at (k0,tn).
// lane holds column n=lane&15, K = k0 + 8*(lane>>4) + {0..7, 16..23}.
__device__ __forceinline__ v16bf load_b_frag(const float* __restrict__ W, int K,
                                             int tn, int k0, int lane) {
  int n  = tn + (lane & 15);
  int kb = k0 + ((lane >> 4) << 3);
  const float4* p = (const float4*)(W + (size_t)n * K + kb);
  float4 q0 = p[0], q1 = p[1], q2 = p[4], q3 = p[5];
  v16bf f;
  f[0]=(__bf16)q0.x; f[1]=(__bf16)q0.y; f[2]=(__bf16)q0.z; f[3]=(__bf16)q0.w;
  f[4]=(__bf16)q1.x; f[5]=(__bf16)q1.y; f[6]=(__bf16)q1.z; f[7]=(__bf16)q1.w;
  f[8]=(__bf16)q2.x; f[9]=(__bf16)q2.y; f[10]=(__bf16)q2.z; f[11]=(__bf16)q2.w;
  f[12]=(__bf16)q3.x; f[13]=(__bf16)q3.y; f[14]=(__bf16)q3.z; f[15]=(__bf16)q3.w;
  return f;
}

#define WMMA_BF16(a,b,acc) \
  __builtin_amdgcn_wmma_f32_16x16x32_bf16(false,(a),false,(b),(short)0,(acc),false,false)

// ---------------- GEMV: hub input projection ----------------
__global__ void k_hub_inproj(const float* __restrict__ x,
                             const float* __restrict__ Wr, const float* __restrict__ br,
                             const float* __restrict__ Wi, const float* __restrict__ bi,
                             float* __restrict__ xpr, float* __restrict__ xpi,
                             float* __restrict__ xmag) {
  int lane = threadIdx.x & 31;
  int row  = blockIdx.x * (blockDim.x >> 5) + (threadIdx.x >> 5);
  if (row >= HH) return;
  const float* wr = Wr + (size_t)row * INN;
  const float* wi = Wi + (size_t)row * INN;
  float sr = 0.f, si = 0.f;
  for (int k = lane; k < INN; k += 32) { float xv = x[k]; sr += xv * wr[k]; si += xv * wi[k]; }
  sr = wave_sum(sr); si = wave_sum(si);
  if (lane == 0) {
    float pr = sr + br[row] - bi[row];
    float pi = si + bi[row] + br[row];
    xpr[row] = pr; xpi[row] = pi; xmag[row] = sqrtf(pr * pr + pi * pi);
  }
}

// ---------------- phase rotation + comb_mag (bf16) ----------------
__global__ void k_phase(const float* __restrict__ hr, const float* __restrict__ hi,
                        const float* __restrict__ pv, const float* __restrict__ xmag,
                        float* __restrict__ sr, float* __restrict__ si,
                        __bf16* __restrict__ combmag) {
  int idx = blockIdx.x * blockDim.x + threadIdx.x;
  if (idx >= NH * HH) return;
  int i = idx >> 11, h = idx & (HH - 1);
  float a = pv[idx] * 0.1f;
  float c = cosf(a), s = sinf(a);
  float rr = hr[idx] * c - hi[idx] * s;
  float ii = hr[idx] * s + hi[idx] * c;
  sr[idx] = rr; si[idx] = ii;
  combmag[(size_t)i * (2 * HH) + h]      = (__bf16)xmag[h];
  combmag[(size_t)i * (2 * HH) + HH + h] = (__bf16)sqrtf(rr * rr + ii * ii);
}

// ---------------- M-strip dual WMMA GEMM: z,r = sigmoid([64,4096] @ {Wz,Wr}^T + b) -------
// One wave per 16-wide N strip; covers all 4 M-tiles so Wz/Wr stream from HBM exactly once.
__global__ void k_gemm_zr(const __bf16* __restrict__ A,
                          const float* __restrict__ B0, const float* __restrict__ b0,
                          const float* __restrict__ B1, const float* __restrict__ b1,
                          float* __restrict__ Z, float* __restrict__ R) {
  const int N = HH, K = 2 * HH;
  int lane = threadIdx.x & 31;
  int wave = blockIdx.x * (blockDim.x >> 5) + (threadIdx.x >> 5);
  if (wave >= (N >> 4)) return;
  int tn = wave << 4;
  v8f az0 = {}, az1 = {}, az2 = {}, az3 = {};
  v8f ar0 = {}, ar1 = {}, ar2 = {}, ar3 = {};
  for (int k0 = 0; k0 < K; k0 += 32) {
    v16bf f0 = load_b_frag(B0, K, tn, k0, lane);
    v16bf f1 = load_b_frag(B1, K, tn, k0, lane);
    v16bf a0 = load_a_frag(A, K,  0, k0, lane);
    v16bf a1 = load_a_frag(A, K, 16, k0, lane);
    v16bf a2 = load_a_frag(A, K, 32, k0, lane);
    v16bf a3 = load_a_frag(A, K, 48, k0, lane);
    az0 = WMMA_BF16(a0, f0, az0);  ar0 = WMMA_BF16(a0, f1, ar0);
    az1 = WMMA_BF16(a1, f0, az1);  ar1 = WMMA_BF16(a1, f1, ar1);
    az2 = WMMA_BF16(a2, f0, az2);  ar2 = WMMA_BF16(a2, f1, ar2);
    az3 = WMMA_BF16(a3, f0, az3);  ar3 = WMMA_BF16(a3, f1, ar3);
  }
  int n = tn + (lane & 15);
  int mr = (lane >> 4) << 3;
  float bb0 = b0[n], bb1 = b1[n];
  v8f az[4] = {az0, az1, az2, az3};
  v8f ar[4] = {ar0, ar1, ar2, ar3};
  #pragma unroll
  for (int t = 0; t < 4; ++t) {
    #pragma unroll
    for (int i = 0; i < 8; ++i) {
      size_t o = (size_t)(t * 16 + mr + i) * N + n;
      Z[o] = sigm(az[t][i] + bb0);
      R[o] = sigm(ar[t][i] + bb1);
    }
  }
}

// ---------------- build cat inputs for cand GEMMs ----------------
__global__ void k_cat(const float* __restrict__ xpr, const float* __restrict__ xpi,
                      const float* __restrict__ r,
                      const float* __restrict__ sr, const float* __restrict__ si,
                      __bf16* __restrict__ catR, __bf16* __restrict__ catI) {
  int idx = blockIdx.x * blockDim.x + threadIdx.x;
  if (idx >= NH * HH) return;
  int i = idx >> 11, h = idx & (HH - 1);
  size_t base = (size_t)i * (2 * HH);
  float rv = r[idx];
  catR[base + h]      = (__bf16)xpr[h];
  catR[base + HH + h] = (__bf16)(rv * sr[idx]);
  catI[base + h]      = (__bf16)xpi[h];
  catI[base + HH + h] = (__bf16)(rv * si[idx]);
}

// ---------------- M-strip dual WMMA GEMM: cand_r/cand_i = tanh(cat @ Wh^T + b) ----------
// Fused: catR@Whr and catI@Whi share the wave; Whr/Whi each stream exactly once.
__global__ void k_gemm_cand(const __bf16* __restrict__ AR, const __bf16* __restrict__ AI,
                            const float* __restrict__ B0, const float* __restrict__ b0,
                            const float* __restrict__ B1, const float* __restrict__ b1,
                            float* __restrict__ CR, float* __restrict__ CI) {
  const int N = HH, K = 2 * HH;
  int lane = threadIdx.x & 31;
  int wave = blockIdx.x * (blockDim.x >> 5) + (threadIdx.x >> 5);
  if (wave >= (N >> 4)) return;
  int tn = wave << 4;
  v8f cr0 = {}, cr1 = {}, cr2 = {}, cr3 = {};
  v8f ci0 = {}, ci1 = {}, ci2 = {}, ci3 = {};
  for (int k0 = 0; k0 < K; k0 += 32) {
    v16bf f0 = load_b_frag(B0, K, tn, k0, lane);
    v16bf f1 = load_b_frag(B1, K, tn, k0, lane);
    v16bf r0 = load_a_frag(AR, K,  0, k0, lane);
    v16bf r1 = load_a_frag(AR, K, 16, k0, lane);
    v16bf r2 = load_a_frag(AR, K, 32, k0, lane);
    v16bf r3 = load_a_frag(AR, K, 48, k0, lane);
    cr0 = WMMA_BF16(r0, f0, cr0);
    cr1 = WMMA_BF16(r1, f0, cr1);
    cr2 = WMMA_BF16(r2, f0, cr2);
    cr3 = WMMA_BF16(r3, f0, cr3);
    v16bf i0 = load_a_frag(AI, K,  0, k0, lane);
    v16bf i1 = load_a_frag(AI, K, 16, k0, lane);
    v16bf i2 = load_a_frag(AI, K, 32, k0, lane);
    v16bf i3 = load_a_frag(AI, K, 48, k0, lane);
    ci0 = WMMA_BF16(i0, f1, ci0);
    ci1 = WMMA_BF16(i1, f1, ci1);
    ci2 = WMMA_BF16(i2, f1, ci2);
    ci3 = WMMA_BF16(i3, f1, ci3);
  }
  int n = tn + (lane & 15);
  int mr = (lane >> 4) << 3;
  float bb0 = b0[n], bb1 = b1[n];
  v8f cr[4] = {cr0, cr1, cr2, cr3};
  v8f ci[4] = {ci0, ci1, ci2, ci3};
  #pragma unroll
  for (int t = 0; t < 4; ++t) {
    #pragma unroll
    for (int i = 0; i < 8; ++i) {
      size_t o = (size_t)(t * 16 + mr + i) * N + n;
      CR[o] = tanh_fast(cr[t][i] + bb0);
      CI[o] = tanh_fast(ci[t][i] + bb1);
    }
  }
}

// ---------------- hub state update + coupling + amplitude ----------------
__global__ void k_hubstate(const float* __restrict__ z,
                           const float* __restrict__ candr, const float* __restrict__ candi,
                           const float* __restrict__ sr, const float* __restrict__ si,
                           const float* __restrict__ hr, const float* __restrict__ hi,
                           float* __restrict__ amp) {
  int idx = blockIdx.x * blockDim.x + threadIdx.x;
  if (idx >= NH * HH) return;
  int i = idx >> 11, h = idx & (HH - 1);
  float zz = z[idx];
  float nr = (1.f - zz) * sr[idx] + zz * candr[idx];
  float ni = (1.f - zz) * si[idx] + zz * candi[idx];
  int ip = ((i + 1) & (NH - 1)) * HH + h, im = ((i + NH - 1) & (NH - 1)) * HH + h;
  float nbr = 0.5f * (hr[ip] + hr[im]);
  float nbi = 0.5f * (hi[ip] + hi[im]);
  float coup = cosf(atan2f(si[idx], sr[idx]) - atan2f(nbi, nbr));
  float str = 0.85f * nr + 0.15f * (coup * nbr * 0.15f);
  float sti = 0.85f * ni + 0.15f * (coup * nbi * 0.15f);
  amp[idx] = sqrtf(str * str + sti * sti);
}

__global__ void k_rowmax(const float* __restrict__ amp, float* __restrict__ ampmax) {
  int i = blockIdx.x;
  float m = -1e30f;
  for (int h = threadIdx.x; h < HH; h += blockDim.x) m = fmaxf(m, amp[(size_t)i * HH + h]);
  m = wave_max(m);
  __shared__ float sm[8];
  if ((threadIdx.x & 31) == 0) sm[threadIdx.x >> 5] = m;
  __syncthreads();
  if (threadIdx.x == 0) {
    float t = sm[0];
    for (int w = 1; w < 8; ++w) t = fmaxf(t, sm[w]);
    ampmax[i] = t;
  }
}

__global__ void k_hubmean(const float* __restrict__ amp, const float* __restrict__ ampmax,
                          float* __restrict__ hmean) {
  int h = blockIdx.x * blockDim.x + threadIdx.x;
  if (h >= HH) return;
  float s = 0.f;
  for (int i = 0; i < NH; ++i) s += amp[(size_t)i * HH + h] * __builtin_amdgcn_rcpf(ampmax[i] + 1e-8f);
  hmean[h] = s * (1.0f / NH);
}

// ---------------- gate GEMV ----------------
__global__ void k_gate(const float* __restrict__ hmean, const float* __restrict__ W,
                       const float* __restrict__ b, float* __restrict__ gate) {
  int lane = threadIdx.x & 31;
  int row  = blockIdx.x * (blockDim.x >> 5) + (threadIdx.x >> 5);
  if (row >= NC) return;
  const float* w = W + (size_t)row * HH;
  float s = 0.f;
  for (int k = lane; k < HH; k += 32) s += hmean[k] * w[k];
  s = wave_sum(s);
  if (lane == 0) gate[row] = sigm(s + b[row]);
}

// ---------------- cortex comb (bf16) ----------------
__global__ void k_comb(const float* __restrict__ x, const float* __restrict__ hid,
                       __bf16* __restrict__ comb) {
  size_t idx = (size_t)blockIdx.x * blockDim.x + threadIdx.x;
  if (idx >= (size_t)NC * (INN + HH)) return;
  int c = (int)(idx / (INN + HH));
  int k = (int)(idx % (INN + HH));
  float v = (k < INN) ? x[k] : hid[(size_t)c * HH + (k - INN)];
  comb[idx] = (__bf16)v;
}

// ---------------- fused dual WMMA: h1{a,g} = relu(comb @ {ea1,eg1}^T + b), bf16 out ------
// ea1/eg1 are 1.5MB each -> WGP$/L2 resident; per-tile form is fine.
__global__ void k_gemm_ea(const __bf16* __restrict__ A,
                          const float* __restrict__ B0, const float* __restrict__ b0,
                          const float* __restrict__ B1, const float* __restrict__ b1,
                          __bf16* __restrict__ H0, __bf16* __restrict__ H1) {
  const int M = NC, N = BM, K = INN + HH;
  int lane = threadIdx.x & 31;
  int wave = blockIdx.x * (blockDim.x >> 5) + (threadIdx.x >> 5);
  int ntiles = N >> 4;
  if (wave >= (M >> 4) * ntiles) return;
  int tm = (wave / ntiles) << 4, tn = (wave % ntiles) << 4;
  v8f acc0 = {}, acc1 = {};
  for (int k0 = 0; k0 < K; k0 += 32) {
    v16bf a  = load_a_frag(A, K, tm, k0, lane);
    v16bf f0 = load_b_frag(B0, K, tn, k0, lane);
    v16bf f1 = load_b_frag(B1, K, tn, k0, lane);
    acc0 = WMMA_BF16(a, f0, acc0);
    acc1 = WMMA_BF16(a, f1, acc1);
  }
  int n = tn + (lane & 15);
  int mr = tm + ((lane >> 4) << 3);
  float bb0 = b0[n], bb1 = b1[n];
  #pragma unroll
  for (int i = 0; i < 8; ++i) {
    H0[(size_t)(mr + i) * N + n] = (__bf16)fmaxf(acc0[i] + bb0, 0.f);
    H1[(size_t)(mr + i) * N + n] = (__bf16)fmaxf(acc1[i] + bb1, 0.f);
  }
}

// ---------------- fused dual WMMA: d = (h1a@ea2^T+b) - (h1g@eg2^T+b) ----------------
__global__ void k_gemm_out(const __bf16* __restrict__ A0, const __bf16* __restrict__ A1,
                           const float* __restrict__ B0, const float* __restrict__ b0,
                           const float* __restrict__ B1, const float* __restrict__ b1,
                           float* __restrict__ D) {
  const int M = NC, N = OUTN, K = BM;
  int lane = threadIdx.x & 31;
  int wave = blockIdx.x * (blockDim.x >> 5) + (threadIdx.x >> 5);
  int ntiles = N >> 4;
  if (wave >= (M >> 4) * ntiles) return;
  int tm = (wave / ntiles) << 4, tn = (wave % ntiles) << 4;
  v8f acc0 = {}, acc1 = {};
  #pragma unroll
  for (int k0 = 0; k0 < K; k0 += 32) {
    v16bf a0 = load_a_frag(A0, K, tm, k0, lane);
    v16bf a1 = load_a_frag(A1, K, tm, k0, lane);
    v16bf f0 = load_b_frag(B0, K, tn, k0, lane);
    v16bf f1 = load_b_frag(B1, K, tn, k0, lane);
    acc0 = WMMA_BF16(a0, f0, acc0);
    acc1 = WMMA_BF16(a1, f1, acc1);
  }
  int n = tn + (lane & 15);
  int mr = tm + ((lane >> 4) << 3);
  float bb = b0[n] - b1[n];
  #pragma unroll
  for (int i = 0; i < 8; ++i)
    D[(size_t)(mr + i) * N + n] = (acc0[i] - acc1[i]) + bb;
}

// ---------------- tension per cortex cell ----------------
__global__ void k_tension(const float* __restrict__ D, const float* __restrict__ gate,
                          float* __restrict__ tens) {
  int c = blockIdx.x;
  const float* row = D + (size_t)c * OUTN;
  float s = 0.f;
  for (int i = threadIdx.x; i < OUTN; i += blockDim.x) { float v = row[i]; s += v * v; }
  s = wave_sum(s);
  __shared__ float sm[8];
  if ((threadIdx.x & 31) == 0) sm[threadIdx.x >> 5] = s;
  __syncthreads();
  if (threadIdx.x == 0) {
    float t = 0.f;
    for (int w = 0; w < 8; ++w) t += sm[w];
    tens[c] = t * (1.0f / OUTN) * gate[c];
  }
}

// ---------------- softmax(tension) * gate -> coeff ----------------
__global__ void k_softmax(const float* __restrict__ tens, const float* __restrict__ gate,
                          float* __restrict__ coeff) {
  __shared__ float sm[256];
  float m = -1e30f;
  for (int i = threadIdx.x; i < NC; i += 256) m = fmaxf(m, tens[i]);
  sm[threadIdx.x] = m; __syncthreads();
  for (int s = 128; s > 0; s >>= 1) {
    if (threadIdx.x < s) sm[threadIdx.x] = fmaxf(sm[threadIdx.x], sm[threadIdx.x + s]);
    __syncthreads();
  }
  float mx = sm[0]; __syncthreads();
  float acc = 0.f;
  for (int i = threadIdx.x; i < NC; i += 256) acc += __expf(tens[i] - mx);
  sm[threadIdx.x] = acc; __syncthreads();
  for (int s = 128; s > 0; s >>= 1) {
    if (threadIdx.x < s) sm[threadIdx.x] += sm[threadIdx.x + s];
    __syncthreads();
  }
  float inv = 1.0f / sm[0];
  for (int i = threadIdx.x; i < NC; i += 256)
    coeff[i] = __expf(tens[i] - mx) * inv * gate[i];
}

// ---------------- weighted column sum: cortex_combined ----------------
__global__ void k_colsum(const float* __restrict__ D, const float* __restrict__ coeff,
                         float* __restrict__ ccomb) {
  int n = blockIdx.x * blockDim.x + threadIdx.x;
  if (n >= OUTN) return;
  float s = 0.f;
  for (int c = 0; c < NC; ++c) s += coeff[c] * D[(size_t)c * OUTN + n];
  ccomb[n] = s;
}

// ---------------- head GEMV ----------------
__global__ void k_head(const float* __restrict__ ccomb, const float* __restrict__ hmean,
                       const float* __restrict__ W, const float* __restrict__ b,
                       float* __restrict__ out) {
  int lane = threadIdx.x & 31;
  int row  = blockIdx.x * (blockDim.x >> 5) + (threadIdx.x >> 5);
  if (row >= INN) return;
  const float* w = W + (size_t)row * (OUTN + HH);
  float s = 0.f;
  for (int k = lane; k < OUTN + HH; k += 32) {
    float cv = (k < OUTN) ? ccomb[k] : hmean[k - OUTN];
    s += cv * w[k];
  }
  s = wave_sum(s);
  if (lane == 0) out[row] = s + b[row];
}

// ---------------- launch ----------------
extern "C" void kernel_launch(void* const* d_in, const int* in_sizes, int n_in,
                              void* d_out, int out_size, void* d_ws, size_t ws_size,
                              hipStream_t stream) {
  const float* x       = (const float*)d_in[0];
  const float* hub_r   = (const float*)d_in[1];
  const float* hub_i   = (const float*)d_in[2];
  const float* pv      = (const float*)d_in[3];
  const float* chid    = (const float*)d_in[4];
  const float* WrW     = (const float*)d_in[5];  const float* WrB = (const float*)d_in[6];
  const float* WiW     = (const float*)d_in[7];  const float* WiB = (const float*)d_in[8];
  const float* WzW     = (const float*)d_in[9];  const float* WzB = (const float*)d_in[10];
  const float* WrrW    = (const float*)d_in[11]; const float* WrrB = (const float*)d_in[12];
  const float* WhrW    = (const float*)d_in[13]; const float* WhrB = (const float*)d_in[14];
  const float* WhiW    = (const float*)d_in[15]; const float* WhiB = (const float*)d_in[16];
  const float* gateW   = (const float*)d_in[17]; const float* gateB = (const float*)d_in[18];
  const float* ea1W    = (const float*)d_in[19]; const float* ea1B = (const float*)d_in[20];
  const float* ea2W    = (const float*)d_in[21]; const float* ea2B = (const float*)d_in[22];
  const float* eg1W    = (const float*)d_in[23]; const float* eg1B = (const float*)d_in[24];
  const float* eg2W    = (const float*)d_in[25]; const float* eg2B = (const float*)d_in[26];
  const float* headW   = (const float*)d_in[27]; const float* headB = (const float*)d_in[28];
  float* out = (float*)d_out;

  // workspace carve-up (f32 region then 16B-aligned bf16 region)
  float* ws    = (float*)d_ws;
  float* xpr   = ws;                 // 2048
  float* xpi   = xpr   + HH;
  float* xmag  = xpi   + HH;
  float* sr    = xmag  + HH;         // 64*2048
  float* si    = sr    + NH * HH;
  float* zbuf  = si    + NH * HH;
  float* rbuf  = zbuf  + NH * HH;
  float* candr = rbuf  + NH * HH;
  float* candi = candr + NH * HH;
  float* amp   = candi + NH * HH;
  float* ampmx = amp   + NH * HH;    // 64
  float* hmean = ampmx + NH;         // 2048
  float* gate  = hmean + HH;         // 1984
  float* dmat  = gate  + NC;         // 1984*1024
  float* tens  = dmat  + (size_t)NC * OUTN;  // 1984
  float* coeff = tens  + NC;         // 1984
  float* ccomb = coeff + NC;         // 1024
  __bf16* combmag = (__bf16*)(ccomb + OUTN);        // 64*4096
  __bf16* catR    = combmag + (size_t)NH * 2 * HH;
  __bf16* catI    = catR    + (size_t)NH * 2 * HH;
  __bf16* comb    = catI    + (size_t)NH * 2 * HH;  // 1984*3072
  __bf16* h1a     = comb    + (size_t)NC * (INN + HH); // 1984*128
  __bf16* h1g     = h1a     + (size_t)NC * BM;

  const int T = 256;

  // 1. hub input projection (GEMV, 2048 rows, wave/row)
  k_hub_inproj<<<HH / 8, T, 0, stream>>>(x, WrW, WrB, WiW, WiB, xpr, xpi, xmag);
  // 2. phase rotation + comb_mag(bf16)
  k_phase<<<(NH * HH) / T, T, 0, stream>>>(hub_r, hub_i, pv, xmag, sr, si, combmag);
  // 3. z,r = sigmoid(dual WMMA, full-M strip)  [64,4096]x[4096,2048]x2
  k_gemm_zr<<<(HH / 16) / 8, T, 0, stream>>>(combmag, WzW, WzB, WrrW, WrrB, zbuf, rbuf);
  // 4. cat inputs for candidates
  k_cat<<<(NH * HH) / T, T, 0, stream>>>(xpr, xpi, rbuf, sr, si, catR, catI);
  // 5. cand_r/cand_i = tanh(dual WMMA, full-M strip)
  k_gemm_cand<<<(HH / 16) / 8, T, 0, stream>>>(catR, catI, WhrW, WhrB, WhiW, WhiB, candr, candi);
  // 6. hub state update + ring coupling + amplitude
  k_hubstate<<<(NH * HH) / T, T, 0, stream>>>(zbuf, candr, candi, sr, si, hub_r, hub_i, amp);
  // 7-8. per-hub max, hub_mean
  k_rowmax<<<NH, T, 0, stream>>>(amp, ampmx);
  k_hubmean<<<HH / T, T, 0, stream>>>(amp, ampmx, hmean);
  // 9. gate GEMV (1984 rows)
  k_gate<<<NC / 8, T, 0, stream>>>(hmean, gateW, gateB, gate);
  // 10. cortex comb (bf16)
  k_comb<<<(int)(((size_t)NC * (INN + HH) + T - 1) / T), T, 0, stream>>>(x, chid, comb);
  // 11. h1a/h1g = relu(dual WMMA) [1984,3072]x[3072,128]x2
  k_gemm_ea<<<(124 * 8 + 7) / 8, T, 0, stream>>>(comb, ea1W, ea1B, eg1W, eg1B, h1a, h1g);
  // 12. d = a - g (dual WMMA)     [1984,128]x[128,1024]x2
  k_gemm_out<<<(124 * 64 + 7) / 8, T, 0, stream>>>(h1a, h1g, ea2W, ea2B, eg2W, eg2B, dmat);
  // 13-15. tension, softmax*gate coeff, weighted column sum
  k_tension<<<NC, T, 0, stream>>>(dmat, gate, tens);
  k_softmax<<<1, T, 0, stream>>>(tens, gate, coeff);
  k_colsum<<<OUTN / T, T, 0, stream>>>(dmat, coeff, ccomb);
  // 16. head GEMV (1024 rows)
  k_head<<<INN / 8, T, 0, stream>>>(ccomb, hmean, headW, headB, out);
}
